// EventDrivenComputeEngine_33071248179949
// MI455X (gfx1250) — matrix operand order repaired and checked by reading, not compiled, not run on hardware
//
#include <hip/hip_runtime.h>
#include <hip/hip_bf16.h>

// ---------------------------------------------------------------------------
// Fused FFN: y = relu(x@W1 + b1)@W2 + b2, spike-masked per row.
// x: [32768, 512] f32, W1: [512,2048], W2: [2048,512].
//
//   * bf16 hi/lo split (3 WMMA passes) => ~fp32 accuracy on
//     v_wmma_f32_16x16x32_bf16 (3/8 the cost of fp32 WMMA on CDNA5).
//   * Weights converted + swizzled into WMMA B-fragment layout once per
//     launch (prep kernel -> d_ws): GEMM inner loops are pure b128 loads
//     + WMMA, zero conversion VALU.
//   * 64-row tiles, 16 waves: each B fragment feeds 12 WMMAs; total L2
//     weight traffic = 512 WGs x 8 MB = 4 GB (~50 TB/s sustained).
// ---------------------------------------------------------------------------

typedef __attribute__((ext_vector_type(16))) __bf16 v16bf;
typedef __attribute__((ext_vector_type(8)))  float  v8f;

union BF16x16 {
    unsigned int u[8];
    v16bf v;
};

#define DIM_D 512
#define DIM_F 2048
#define ROWS_TOTAL 32768
#define MT 64            // rows per workgroup (4 WMMA M-tiles per wave)
#define FC 256           // F-chunk width (16 waves x 16 cols)
#define THREADS 512      // 16 waves (wave32)

// plane sizes in dwords (2 bf16 per dword)
#define W1_DWORDS (DIM_D * DIM_F / 2)   // 524288
#define W2_DWORDS (DIM_F * DIM_D / 2)   // 524288

__device__ __forceinline__ unsigned short bf16_rne(float f) {
    unsigned int u = __float_as_uint(f);
    unsigned int r = 0x7FFFu + ((u >> 16) & 1u);   // round-to-nearest-even
    return (unsigned short)((u + r) >> 16);
}
__device__ __forceinline__ float bf16_f(unsigned short h) {
    return __uint_as_float(((unsigned int)h) << 16);
}
__device__ __forceinline__ void bf16_split(float f, unsigned short& hi, unsigned short& lo) {
    hi = bf16_rne(f);
    lo = bf16_rne(f - bf16_f(hi));
}

// 3-pass hi/lo bf16 WMMA: c += a*b with ~fp32 accuracy
__device__ __forceinline__ v8f wmma3(const BF16x16& ah, const BF16x16& al,
                                     const BF16x16& bh, const BF16x16& bl, v8f c) {
    c = __builtin_amdgcn_wmma_f32_16x16x32_bf16(false, ah.v, false, bh.v, (short)0, c, false, false);
    c = __builtin_amdgcn_wmma_f32_16x16x32_bf16(false, ah.v, false, bl.v, (short)0, c, false, false);
    c = __builtin_amdgcn_wmma_f32_16x16x32_bf16(false, al.v, false, bh.v, (short)0, c, false, false);
    return c;
}

// ---------------------------------------------------------------------------
// Weight prep: convert f32 W[K][N] into bf16 hi/lo planes swizzled to the
// WMMA B-fragment layout. Per 32x16 (KxN) tile: 32 lanes x 8 dwords, where
// lane l covers col=(l&15), K-half=(l&16)?16:0, dword j packs K=2j (lo16)
// and K=2j+1 (hi16). A fragment load is then 2 contiguous b128 per plane.
// ---------------------------------------------------------------------------
__global__ __launch_bounds__(256)
void prep_weights_kernel(const float* __restrict__ W, int K, int N,
                         unsigned int* __restrict__ hi_plane,
                         unsigned int* __restrict__ lo_plane) {
    int g = blockIdx.x * blockDim.x + threadIdx.x;
    int tiles_n = N >> 4;
    int total   = (K >> 5) * tiles_n * 32;
    if (g >= total) return;
    int lane = g & 31;
    int tile = g >> 5;
    int kb   = tile / tiles_n;
    int nb   = tile - kb * tiles_n;
    int col   = (nb << 4) + (lane & 15);
    int kbase = (kb << 5) + ((lane & 16) ? 16 : 0);

    unsigned int hu[8], lu[8];
#pragma unroll
    for (int j = 0; j < 8; ++j) {
        float f0 = W[(size_t)(kbase + 2 * j)     * N + col];
        float f1 = W[(size_t)(kbase + 2 * j + 1) * N + col];
        unsigned short h0, l0, h1, l1;
        bf16_split(f0, h0, l0);
        bf16_split(f1, h1, l1);
        hu[j] = (unsigned int)h0 | ((unsigned int)h1 << 16);
        lu[j] = (unsigned int)l0 | ((unsigned int)l1 << 16);
    }
    size_t base = (size_t)g * 8;
    *(uint4*)(hi_plane + base)     = make_uint4(hu[0], hu[1], hu[2], hu[3]);
    *(uint4*)(hi_plane + base + 4) = make_uint4(hu[4], hu[5], hu[6], hu[7]);
    *(uint4*)(lo_plane + base)     = make_uint4(lu[0], lu[1], lu[2], lu[3]);
    *(uint4*)(lo_plane + base + 4) = make_uint4(lu[4], lu[5], lu[6], lu[7]);
}

// B fragment from a swizzled plane: two contiguous 16B loads, no VALU.
__device__ __forceinline__ void load_b_sw(const unsigned int* __restrict__ plane,
                                          int tiles_n, int k0, int col, int lane,
                                          BF16x16& out) {
    size_t base = ((size_t)((k0 >> 5) * tiles_n + (col >> 4)) * 32 + lane) * 8;
    uint4 a = *(const uint4*)(plane + base);
    uint4 b = *(const uint4*)(plane + base + 4);
    out.u[0] = a.x; out.u[1] = a.y; out.u[2] = a.z; out.u[3] = a.w;
    out.u[4] = b.x; out.u[5] = b.y; out.u[6] = b.z; out.u[7] = b.w;
}

// A fragment (16x32 bf16) from an LDS plane laid out [row][k] (k contiguous).
// Contiguous-K order (matches B planes): lanes 0-15: row=lane, K {0..15}; lanes 16-31: row=lane-16,
// K {16..31}. Two aligned 16B LDS reads.
__device__ __forceinline__ void load_a_frag(const unsigned short* plane, int rs,
                                            int k0, int lane, BF16x16& out) {
    int m  = lane & 15;
    int ko = (lane & 16) ? 16 : 0;
    const unsigned short* p = plane + m * rs + k0 + ko;
    uint4 a = *(const uint4*)(p);        // K = k0+ko    .. +7
    uint4 b = *(const uint4*)(p + 8);    // K = k0+ko+8  .. +15
    out.u[0] = a.x; out.u[1] = a.y; out.u[2] = a.z; out.u[3] = a.w;
    out.u[4] = b.x; out.u[5] = b.y; out.u[6] = b.z; out.u[7] = b.w;
}

__global__ __launch_bounds__(THREADS)
void ffn_spike_wmma_kernel(const float* __restrict__ x,
                           const float* __restrict__ b1,
                           const float* __restrict__ b2,
                           const unsigned int* __restrict__ w1h,
                           const unsigned int* __restrict__ w1l,
                           const unsigned int* __restrict__ w2h,
                           const unsigned int* __restrict__ w2l,
                           float* __restrict__ out) {
    __shared__ __align__(16) unsigned short xs_hi[MT * DIM_D];   // 64 KB
    __shared__ __align__(16) unsigned short xs_lo[MT * DIM_D];   // 64 KB
    __shared__ __align__(16) unsigned short hs_hi[MT * FC];      // 32 KB
    __shared__ __align__(16) unsigned short hs_lo[MT * FC];      // 32 KB
    __shared__ int rowmask[MT];

    const int tid  = threadIdx.x;
    const int lane = tid & 31;
    const int wv   = tid >> 5;                // wave 0..15
    const int r0   = blockIdx.x * MT;         // first row of this tile

    if (tid < MT) rowmask[tid] = 0;
    __syncthreads();

    // ---- Phase 0: stage X tile as bf16 hi/lo planes + spike mask ----------
    {
        int row = tid >> 3;       // 8 threads per row (64 rows, 512 threads)
        int c4  = tid & 7;        // float4 column group
        const float4* src = (const float4*)(x + (size_t)(r0 + row) * DIM_D);
        int active = 0;
#pragma unroll
        for (int j = 0; j < 16; ++j) {
            float4 v = src[c4 + j * 8];
            active |= (fabsf(v.x) > 0.01f) | (fabsf(v.y) > 0.01f) |
                      (fabsf(v.z) > 0.01f) | (fabsf(v.w) > 0.01f);
            int k = (c4 + j * 8) * 4;
            unsigned short h, l;
            bf16_split(v.x, h, l); xs_hi[row * DIM_D + k + 0] = h; xs_lo[row * DIM_D + k + 0] = l;
            bf16_split(v.y, h, l); xs_hi[row * DIM_D + k + 1] = h; xs_lo[row * DIM_D + k + 1] = l;
            bf16_split(v.z, h, l); xs_hi[row * DIM_D + k + 2] = h; xs_lo[row * DIM_D + k + 2] = l;
            bf16_split(v.w, h, l); xs_hi[row * DIM_D + k + 3] = h; xs_lo[row * DIM_D + k + 3] = l;
        }
        if (active) atomicOr(&rowmask[row], 1);
    }
    __syncthreads();

    // Output accumulators: wave covers 32 output cols x 64 rows = 2x4 tiles.
    const int d0 = wv * 32;
    v8f acc[8];
#pragma unroll
    for (int t = 0; t < 8; ++t) acc[t] = (v8f){0.f,0.f,0.f,0.f,0.f,0.f,0.f,0.f};

    // ---- Main loop over F in chunks of 256 --------------------------------
#pragma unroll 1
    for (int fc = 0; fc < DIM_F; fc += FC) {
        // GEMM1: this wave computes h[0:64, fc+wv*16 : +16] (4 M-tiles)
        const int col1 = fc + wv * 16;
        v8f hc[4];
#pragma unroll
        for (int i = 0; i < 4; ++i) hc[i] = (v8f){0.f,0.f,0.f,0.f,0.f,0.f,0.f,0.f};

#pragma unroll 1
        for (int k0 = 0; k0 < DIM_D; k0 += 32) {
            BF16x16 bh, bl;
            load_b_sw(w1h, DIM_F >> 4, k0, col1, lane, bh);
            load_b_sw(w1l, DIM_F >> 4, k0, col1, lane, bl);
#pragma unroll
            for (int mh = 0; mh < 4; ++mh) {
                BF16x16 ah, al;
                load_a_frag(xs_hi + mh * 16 * DIM_D, DIM_D, k0, lane, ah);
                load_a_frag(xs_lo + mh * 16 * DIM_D, DIM_D, k0, lane, al);
                hc[mh] = wmma3(ah, al, bh, bl, hc[mh]);
            }
        }

        float b1v = b1[col1 + (lane & 15)];

        __syncthreads();   // previous chunk's h readers are done (WAR)

        // bias + ReLU + split h tiles into LDS hi/lo planes [m][k_local]
        {
            int n  = wv * 16 + (lane & 15);
            int mb = (lane & 16) ? 8 : 0;
#pragma unroll
            for (int mh = 0; mh < 4; ++mh) {
#pragma unroll
                for (int i = 0; i < 8; ++i) {
                    float v = hc[mh][i] + b1v;
                    v = v > 0.f ? v : 0.f;
                    unsigned short h, l;
                    bf16_split(v, h, l);
                    hs_hi[(mh * 16 + mb + i) * FC + n] = h;
                    hs_lo[(mh * 16 + mb + i) * FC + n] = l;
                }
            }
        }
        __syncthreads();   // h planes complete

        // GEMM2: acc += h[64 x FC] @ W2[fc:fc+FC, d0:d0+32]
#pragma unroll 1
        for (int k2 = 0; k2 < FC; k2 += 32) {
            BF16x16 bh0, bl0, bh1, bl1;
            load_b_sw(w2h, DIM_D >> 4, fc + k2, d0,      lane, bh0);
            load_b_sw(w2l, DIM_D >> 4, fc + k2, d0,      lane, bl0);
            load_b_sw(w2h, DIM_D >> 4, fc + k2, d0 + 16, lane, bh1);
            load_b_sw(w2l, DIM_D >> 4, fc + k2, d0 + 16, lane, bl1);
#pragma unroll
            for (int mh = 0; mh < 4; ++mh) {
                BF16x16 ah, al;
                load_a_frag(hs_hi + mh * 16 * FC, FC, k2, lane, ah);
                load_a_frag(hs_lo + mh * 16 * FC, FC, k2, lane, al);
                acc[mh]     = wmma3(ah, al, bh0, bl0, acc[mh]);
                acc[4 + mh] = wmma3(ah, al, bh1, bl1, acc[4 + mh]);
            }
        }
    }

    // ---- Epilogue: + b2, spike mask, coalesced store ----------------------
    {
        int n  = lane & 15;
        int mb = (lane & 16) ? 8 : 0;
#pragma unroll
        for (int t = 0; t < 2; ++t) {
            int d = d0 + t * 16 + n;
            float bv = b2[d];
#pragma unroll
            for (int mh = 0; mh < 4; ++mh) {
#pragma unroll
                for (int i = 0; i < 8; ++i) {
                    int m = mh * 16 + mb + i;
                    float v = acc[t * 4 + mh][i] + bv;
                    v = rowmask[m] ? v : 0.f;
                    out[(size_t)(r0 + m) * DIM_D + d] = v;
                }
            }
        }
    }
}

extern "C" void kernel_launch(void* const* d_in, const int* in_sizes, int n_in,
                              void* d_out, int out_size, void* d_ws, size_t ws_size,
                              hipStream_t stream) {
    (void)in_sizes; (void)n_in; (void)ws_size; (void)out_size;
    const float* x  = (const float*)d_in[0];
    const float* w1 = (const float*)d_in[1];
    const float* b1 = (const float*)d_in[2];
    const float* w2 = (const float*)d_in[3];
    const float* b2 = (const float*)d_in[4];
    float* out = (float*)d_out;

    unsigned int* ws  = (unsigned int*)d_ws;
    unsigned int* w1h = ws;
    unsigned int* w1l = ws + W1_DWORDS;
    unsigned int* w2h = ws + 2 * (size_t)W1_DWORDS;
    unsigned int* w2l = ws + 2 * (size_t)W1_DWORDS + W2_DWORDS;

    // Prep: swizzle weights into WMMA B-fragment bf16 hi/lo planes (~8 MB).
    {
        int total1 = (DIM_D >> 5) * (DIM_F >> 4) * 32;   // 65536 threads
        prep_weights_kernel<<<(total1 + 255) / 256, 256, 0, stream>>>(
            w1, DIM_D, DIM_F, w1h, w1l);
        int total2 = (DIM_F >> 5) * (DIM_D >> 4) * 32;   // 65536 threads
        prep_weights_kernel<<<(total2 + 255) / 256, 256, 0, stream>>>(
            w2, DIM_F, DIM_D, w2h, w2l);
    }

    dim3 grid(ROWS_TOTAL / MT);   // 512 workgroups of 64 rows
    dim3 block(THREADS);          // 16 waves (wave32)
    ffn_spike_wmma_kernel<<<grid, block, 0, stream>>>(x, b1, b2, w1h, w1l, w2h, w2l, out);
}
